// BGFusionBlock_28097676051139
// MI455X (gfx1250) — matrix operand order repaired and checked by reading, not compiled, not run on hardware
//
#include <hip/hip_runtime.h>

typedef _Float16 f16;
typedef __attribute__((ext_vector_type(16))) _Float16 v16h;
typedef __attribute__((ext_vector_type(8)))  _Float16 v8h;
typedef __attribute__((ext_vector_type(8)))  float    v8f;

union Frag16 { v16h v; v8h h[2]; };

#define DIM_B 4
#define DIM_T 8
#define DIM_C 64
#define DIM_H 128
#define DIM_W 128
#define HW    16384   // 128*128
#define CHW   1048576 // 64*16384

// ---------------------------------------------------------------------------
// Prep: pack conv weights into f16 with K contiguous so that A-fragments are
// two contiguous 8-half chunks per lane (A layout: M=l&15, K=(l>=16?16:0)+...).
// w1/w2: [O=64][I=64][3][3] -> wXh[s][o][i] ; wf: [64][512] -> wfh[t][o][c]
// ---------------------------------------------------------------------------
__global__ void pack_weights_kernel(const float* __restrict__ w1,
                                    const float* __restrict__ w2,
                                    const float* __restrict__ wf,
                                    f16* __restrict__ w1h,
                                    f16* __restrict__ w2h,
                                    f16* __restrict__ wfh) {
  int i = blockIdx.x * 256 + threadIdx.x;
  if (i < 9 * 64 * 64) {
    int s = i >> 12, o = (i >> 6) & 63, c = i & 63;
    w1h[i] = (f16)w1[(o * 64 + c) * 9 + s];
    w2h[i] = (f16)w2[(o * 64 + c) * 9 + s];
  }
  if (i < 8 * 64 * 64) {
    int t = i >> 12, o = (i >> 6) & 63, c = i & 63;
    wfh[i] = (f16)wf[o * 512 + t * 64 + c];
  }
}

// ---------------------------------------------------------------------------
// xsum[b,c,hw] = f16( sum_t x[b,t,c,hw] )   (conv is linear: emb_sum=conv2(xsum)+T*b2)
// ---------------------------------------------------------------------------
__global__ void xsum_kernel(const float* __restrict__ x, f16* __restrict__ xs) {
  int i = blockIdx.x * 256 + threadIdx.x;
  if (i >= DIM_B * CHW) return;
  int b = i / CHW, r = i - b * CHW;
  const float* p = x + (size_t)b * DIM_T * CHW + r;
  float s = 0.f;
#pragma unroll
  for (int t = 0; t < DIM_T; ++t) s += p[t * CHW];
  xs[i] = (f16)s;
}

// ---------------------------------------------------------------------------
// S[b,c,h,w] = conv3x3(xsum_f16, w2) + 8*b2   (emb_sum), via WMMA.
// grid: (64, B), block 256 (8 waves); 16x16 output block, 18x18 halo in LDS.
// LDS activation layout [y][x][c] -> B-fragment = contiguous v16h per lane.
// A-fragments hoisted per shift and reused for both output rows of the wave.
// ---------------------------------------------------------------------------
__global__ __launch_bounds__(256) void conv_sum_kernel(
    const f16* __restrict__ xs, const f16* __restrict__ w2h,
    const float* __restrict__ b2, float* __restrict__ S) {
  __shared__ f16 tile[18 * 20 * 64];
  const int b = blockIdx.y;
  const int ty = blockIdx.x >> 3, tx = blockIdx.x & 7;
  const int h0 = ty * 16, w0 = tx * 16;
  const int tid = threadIdx.x;
  const f16* xb = xs + (size_t)b * CHW;
  for (int i = tid; i < 64 * 18 * 18; i += 256) {
    int xcol = i % 18, rest = i / 18;
    int yrow = rest % 18, c = rest / 18;
    int gh = h0 - 1 + yrow, gw = w0 - 1 + xcol;
    f16 v = (f16)0.f;
    if ((unsigned)gh < 128u && (unsigned)gw < 128u) v = xb[c * HW + gh * DIM_W + gw];
    tile[(yrow * 20 + xcol) * 64 + c] = v;
  }
  __syncthreads();
  const int wave = tid >> 5, lane = tid & 31;
  const int ln = lane & 15, lh = lane >> 4, kb = lh * 16;
  v8f acc[2][4];
#pragma unroll
  for (int g = 0; g < 2; ++g)
#pragma unroll
    for (int mt = 0; mt < 4; ++mt)
#pragma unroll
      for (int r = 0; r < 8; ++r) acc[g][mt][r] = 8.0f * b2[mt * 16 + r + 8 * lh];
  for (int s = 0; s < 9; ++s) {
    const int dy = s / 3, dx = s - 3 * dy;
    Frag16 af[2][4];
#pragma unroll
    for (int kf = 0; kf < 2; ++kf)
#pragma unroll
      for (int mt = 0; mt < 4; ++mt) {
        const f16* ap = w2h + s * 4096 + kf * 32 + kb + (mt * 16 + ln) * 64;
        af[kf][mt].h[0] = *(const v8h*)(ap);
        af[kf][mt].h[1] = *(const v8h*)(ap + 8);
      }
#pragma unroll
    for (int g = 0; g < 2; ++g) {
      const int rr = wave * 2 + g;
#pragma unroll
      for (int kf = 0; kf < 2; ++kf) {
        Frag16 bf_;  // B: lane l -> N=l&15, K = 16*(l>=16)+j : contiguous
        bf_.v = *(const v16h*)&tile[((rr + dy) * 20 + (ln + dx)) * 64 + kf * 32 + lh * 16];
#pragma unroll
        for (int mt = 0; mt < 4; ++mt)
          acc[g][mt] = __builtin_amdgcn_wmma_f32_16x16x32_f16(
              false, af[kf][mt].v, false, bf_.v, (short)0, acc[g][mt], false, false);
      }
    }
  }
#pragma unroll
  for (int g = 0; g < 2; ++g) {
    const int rr = wave * 2 + g;
    const int pix = (h0 + rr) * DIM_W + w0 + ln;
    float* Sp = S + (size_t)b * CHW + pix;
#pragma unroll
    for (int mt = 0; mt < 4; ++mt)
#pragma unroll
      for (int r = 0; r < 8; ++r) Sp[(mt * 16 + r + 8 * lh) * HW] = acc[g][mt][r];
  }
}

// ---------------------------------------------------------------------------
// scores[b,t,h,w] = sum_c (conv3x3(x[b,t], w1)+b1)[c] * S[b,c]  via WMMA conv
// + per-lane dot + cross-half shuffle reduce. embedding_ref never materialized.
// grid: (64, B*T)
// ---------------------------------------------------------------------------
__global__ __launch_bounds__(256) void score_kernel(
    const float* __restrict__ x, const f16* __restrict__ w1h,
    const float* __restrict__ b1, const float* __restrict__ S,
    float* __restrict__ scores) {
  __shared__ f16 tile[18 * 20 * 64];
  const int bt = blockIdx.y;
  const int b = bt >> 3;
  const int ty = blockIdx.x >> 3, tx = blockIdx.x & 7;
  const int h0 = ty * 16, w0 = tx * 16;
  const int tid = threadIdx.x;
  const float* xb = x + (size_t)bt * CHW;
  for (int i = tid; i < 64 * 18 * 18; i += 256) {
    int xcol = i % 18, rest = i / 18;
    int yrow = rest % 18, c = rest / 18;
    int gh = h0 - 1 + yrow, gw = w0 - 1 + xcol;
    float v = 0.f;
    if ((unsigned)gh < 128u && (unsigned)gw < 128u) v = xb[c * HW + gh * DIM_W + gw];
    tile[(yrow * 20 + xcol) * 64 + c] = (f16)v;
  }
  __syncthreads();
  const int wave = tid >> 5, lane = tid & 31;
  const int ln = lane & 15, lh = lane >> 4, kb = lh * 16;
  v8f acc[2][4];
#pragma unroll
  for (int g = 0; g < 2; ++g)
#pragma unroll
    for (int mt = 0; mt < 4; ++mt)
#pragma unroll
      for (int r = 0; r < 8; ++r) acc[g][mt][r] = b1[mt * 16 + r + 8 * lh];
  for (int s = 0; s < 9; ++s) {
    const int dy = s / 3, dx = s - 3 * dy;
    Frag16 af[2][4];
#pragma unroll
    for (int kf = 0; kf < 2; ++kf)
#pragma unroll
      for (int mt = 0; mt < 4; ++mt) {
        const f16* ap = w1h + s * 4096 + kf * 32 + kb + (mt * 16 + ln) * 64;
        af[kf][mt].h[0] = *(const v8h*)(ap);
        af[kf][mt].h[1] = *(const v8h*)(ap + 8);
      }
#pragma unroll
    for (int g = 0; g < 2; ++g) {
      const int rr = wave * 2 + g;
#pragma unroll
      for (int kf = 0; kf < 2; ++kf) {
        Frag16 bf_;
        bf_.v = *(const v16h*)&tile[((rr + dy) * 20 + (ln + dx)) * 64 + kf * 32 + lh * 16];
#pragma unroll
        for (int mt = 0; mt < 4; ++mt)
          acc[g][mt] = __builtin_amdgcn_wmma_f32_16x16x32_f16(
              false, af[kf][mt].v, false, bf_.v, (short)0, acc[g][mt], false, false);
      }
    }
  }
  // dot over channels with emb_sum S; lane holds channels {mt*16+r+8*lh}
#pragma unroll
  for (int g = 0; g < 2; ++g) {
    const int rr = wave * 2 + g;
    const int pix = (h0 + rr) * DIM_W + w0 + ln;
    const float* Sp = S + (size_t)b * CHW + pix;
    float partial = 0.f;
#pragma unroll
    for (int mt = 0; mt < 4; ++mt)
#pragma unroll
      for (int r = 0; r < 8; ++r) partial += acc[g][mt][r] * Sp[(mt * 16 + r + 8 * lh) * HW];
    partial += __shfl_xor(partial, 16, 32);  // combine channel halves (same pixel)
    if (lh == 0) scores[(size_t)bt * HW + pix] = partial;
  }
}

// ---------------------------------------------------------------------------
// attn = softmax over t of scores/0.5
// ---------------------------------------------------------------------------
__global__ void softmax_kernel(const float* __restrict__ scores,
                               float* __restrict__ attn) {
  int i = blockIdx.x * 256 + threadIdx.x;
  if (i >= DIM_B * HW) return;
  int b = i / HW, p = i - b * HW;
  const float* sp = scores + (size_t)b * DIM_T * HW + p;
  float v[DIM_T], m = -1e30f;
#pragma unroll
  for (int t = 0; t < DIM_T; ++t) { v[t] = 2.0f * sp[t * HW]; m = fmaxf(m, v[t]); }
  float sum = 0.f;
#pragma unroll
  for (int t = 0; t < DIM_T; ++t) { v[t] = __expf(v[t] - m); sum += v[t]; }
  float inv = 1.0f / sum;
  float* ap = attn + (size_t)b * DIM_T * HW + p;
#pragma unroll
  for (int t = 0; t < DIM_T; ++t) ap[t * HW] = v[t] * inv;
}

// ---------------------------------------------------------------------------
// out[b,o,p] = LeakyReLU( bf[o] + sum_t attn[b,t,p] * (Wf_t @ x[b,t,:,p])[o] )
// 1x1 conv per t as WMMA GEMM; attn scale applied per-lane (N=pixel is per-lane).
// A-fragments per t hoisted above the row loop; next frame's tile prefetched
// (global_prefetch_b8) while current frame's WMMAs run.
// grid: (64, B), 256-pixel flat tiles, no halo.
// ---------------------------------------------------------------------------
__global__ __launch_bounds__(256) void fuse_kernel(
    const float* __restrict__ x, const f16* __restrict__ wfh,
    const float* __restrict__ bfv, const float* __restrict__ attn,
    float* __restrict__ out) {
  __shared__ f16 tile[256 * 64];
  const int b = blockIdx.y;
  const int p0 = blockIdx.x * 256;
  const int tid = threadIdx.x;
  const int wave = tid >> 5, lane = tid & 31;
  const int ln = lane & 15, lh = lane >> 4, kb = lh * 16;
  v8f facc[4][2];
#pragma unroll
  for (int mt = 0; mt < 4; ++mt)
#pragma unroll
    for (int g = 0; g < 2; ++g)
#pragma unroll
      for (int r = 0; r < 8; ++r) facc[mt][g][r] = 0.f;
  for (int t = 0; t < DIM_T; ++t) {
    __syncthreads();
    const float* xb = x + (size_t)(b * DIM_T + t) * CHW + p0;
    for (int i = tid; i < 64 * 256; i += 256) {
      int p = i & 255, c = i >> 8;
      tile[p * 64 + c] = (f16)xb[c * HW + p];
    }
    __syncthreads();
    if (t + 1 < DIM_T) {  // prefetch next frame's 64KB tile (512 x 128B lines)
      const float* xn = x + (size_t)(b * DIM_T + t + 1) * CHW + p0;
#pragma unroll
      for (int k = 0; k < 2; ++k) {
        int L = tid * 2 + k;
        __builtin_prefetch(xn + (L >> 3) * HW + (L & 7) * 32, 0, 3);
      }
    }
    const f16* wt = wfh + t * 4096;
    Frag16 af[2][4];
#pragma unroll
    for (int kf = 0; kf < 2; ++kf)
#pragma unroll
      for (int mt = 0; mt < 4; ++mt) {
        const f16* ap = wt + (mt * 16 + ln) * 64 + kf * 32 + kb;
        af[kf][mt].h[0] = *(const v8h*)(ap);
        af[kf][mt].h[1] = *(const v8h*)(ap + 8);
      }
#pragma unroll
    for (int g = 0; g < 2; ++g) {
      const int pb = (wave * 2 + g) * 16;
      const float a = attn[(size_t)(b * DIM_T + t) * HW + p0 + pb + ln];
      Frag16 b0, b1;
      const f16* bp = &tile[(pb + ln) * 64 + lh * 16];
      b0.v = *(const v16h*)(bp);        // K-frag 0 (c 0..31)
      b1.v = *(const v16h*)(bp + 32);   // K-frag 1 (c 32..63)
#pragma unroll
      for (int mt = 0; mt < 4; ++mt) {
        v8f gacc = {};
        gacc = __builtin_amdgcn_wmma_f32_16x16x32_f16(false, af[0][mt].v, false, b0.v,
                                                      (short)0, gacc, false, false);
        gacc = __builtin_amdgcn_wmma_f32_16x16x32_f16(false, af[1][mt].v, false, b1.v,
                                                      (short)0, gacc, false, false);
#pragma unroll
        for (int r = 0; r < 8; ++r) facc[mt][g][r] += a * gacc[r];
      }
    }
  }
#pragma unroll
  for (int mt = 0; mt < 4; ++mt)
#pragma unroll
    for (int g = 0; g < 2; ++g) {
      const int pb = (wave * 2 + g) * 16;
#pragma unroll
      for (int r = 0; r < 8; ++r) {
        const int o = mt * 16 + r + 8 * lh;
        float v = facc[mt][g][r] + bfv[o];
        out[(size_t)b * CHW + o * HW + p0 + pb + ln] = (v >= 0.f) ? v : 0.1f * v;
      }
    }
}

// ---------------------------------------------------------------------------
extern "C" void kernel_launch(void* const* d_in, const int* in_sizes, int n_in,
                              void* d_out, int out_size, void* d_ws, size_t ws_size,
                              hipStream_t stream) {
  const float* x  = (const float*)d_in[0];  // [4,8,64,128,128]
  const float* w1 = (const float*)d_in[1];  // [64,64,3,3]
  const float* b1 = (const float*)d_in[2];  // [64]
  const float* w2 = (const float*)d_in[3];  // [64,64,3,3]
  const float* b2 = (const float*)d_in[4];  // [64]
  const float* wf = (const float*)d_in[5];  // [64,512,1,1]
  const float* bf = (const float*)d_in[6];  // [64]
  float* out = (float*)d_out;               // [4,64,128,128]

  char* ws = (char*)d_ws;
  f16*   xsum   = (f16*)(ws);                       //  8,388,608 B
  float* S      = (float*)(ws + 8388608);           // 16,777,216 B
  float* scores = (float*)(ws + 25165824);          //  2,097,152 B
  float* attn   = (float*)(ws + 27262976);          //  2,097,152 B
  f16*   w1h    = (f16*)(ws + 29360128);            //     73,728 B
  f16*   w2h    = (f16*)(ws + 29433856);            //     73,728 B
  f16*   wfh    = (f16*)(ws + 29507584);            //     65,536 B  (total ~29.6 MB)

  pack_weights_kernel<<<(9 * 64 * 64 + 255) / 256, 256, 0, stream>>>(w1, w2, wf,
                                                                     w1h, w2h, wfh);
  xsum_kernel<<<(DIM_B * CHW + 255) / 256, 256, 0, stream>>>(x, xsum);
  conv_sum_kernel<<<dim3(64, DIM_B), 256, 0, stream>>>(xsum, w2h, b2, S);
  score_kernel<<<dim3(64, DIM_B * DIM_T), 256, 0, stream>>>(x, w1h, b1, S, scores);
  softmax_kernel<<<(DIM_B * HW + 255) / 256, 256, 0, stream>>>(scores, attn);
  fuse_kernel<<<dim3(64, DIM_B), 256, 0, stream>>>(x, wfh, bf, attn, out);
}
